// MambaBlock2D_53077205844149
// MI455X (gfx1250) — matrix operand order repaired and checked by reading, not compiled, not run on hardware
//
#include <hip/hip_runtime.h>
#include <hip/hip_bf16.h>

// ---------------------------------------------------------------------------
// MambaBlock2D for MI455X (gfx1250, wave32, WMMA + TDM)
// B=32, C=192, H=W=64 -> L=4096 > 256 path:
//   pool 4x4 -> (32,256,192), LN, in_proj GEMM, conv1d+SiLU, x_proj GEMM,
//   softplus delta, selective scan (L=256), gate+LN, out_proj GEMM,
//   bilinear 16->64 upsample + depthwise 7x7 conv + gated residual.
// GEMMs: v_wmma_f32_16x16x32_bf16; A tile staged to LDS via tensor_load_to_lds.
// ---------------------------------------------------------------------------

typedef __attribute__((ext_vector_type(16))) __bf16       v16bf;
typedef __attribute__((ext_vector_type(8)))  float        v8f;
typedef __attribute__((ext_vector_type(8)))  unsigned int v8u;
typedef __attribute__((ext_vector_type(4)))  unsigned int v4u;

#define BSZ     32
#define CDIM    192
#define HDIM    64
#define WDIM    64
#define LS      256          // downsampled sequence length (16x16)
#define NROW    (BSZ * LS)   // 8192 rows through the SSM
#define DIN     384          // d_inner
#define DXZ     768          // 2*d_inner
#define DST     16           // d_state
#define DBCP    48           // x_proj output padded 33 -> 48 (mult of 16)

__device__ __forceinline__ unsigned short f2bf(float f) {
    unsigned int u = __float_as_uint(f);
    unsigned int r = (u + 0x7FFFu + ((u >> 16) & 1u)) >> 16;   // RNE
    return (unsigned short)r;
}

// ---------------------------------------------------------------------------
// 1) Convert weights to bf16 (x_proj padded 33->48 with zero rows)
// ---------------------------------------------------------------------------
__global__ void convert_weights_kernel(const float* __restrict__ ipw,   // (768,192)
                                       const float* __restrict__ xpw,   // (33,384)
                                       const float* __restrict__ opw,   // (192,384)
                                       unsigned short* __restrict__ ipb,
                                       unsigned short* __restrict__ xpb,
                                       unsigned short* __restrict__ opb) {
    int idx = blockIdx.x * 256 + threadIdx.x;
    const int NIP = DXZ * CDIM;      // 147456
    const int NXP = DBCP * DIN;      // 18432
    const int NOP = CDIM * DIN;      // 73728
    if (idx < NIP) ipb[idx] = f2bf(ipw[idx]);
    if (idx < NXP) {
        int n = idx / DIN, k = idx % DIN;
        xpb[idx] = (n < 33) ? f2bf(xpw[n * DIN + k]) : (unsigned short)0;
    }
    if (idx < NOP) opb[idx] = f2bf(opw[idx]);
}

// ---------------------------------------------------------------------------
// 2) 4x4 average pool + transpose to (B, Ls, C)
// ---------------------------------------------------------------------------
__global__ void pool_kernel(const float* __restrict__ x, float* __restrict__ xf) {
    int idx = blockIdx.x * 256 + threadIdx.x;          // B*Ls*C = 1572864
    if (idx >= BSZ * LS * CDIM) return;
    int c = idx % CDIM;
    int l = (idx / CDIM) % LS;
    int b = idx / (CDIM * LS);
    int hs = l >> 4, ws = l & 15;
    const float* px = x + ((size_t)(b * CDIM + c) * HDIM + hs * 4) * WDIM + ws * 4;
    float s = 0.f;
#pragma unroll
    for (int i = 0; i < 4; ++i)
#pragma unroll
        for (int j = 0; j < 4; ++j) s += px[i * WDIM + j];
    xf[idx] = s * (1.f / 16.f);
}

// ---------------------------------------------------------------------------
// 3) Generic row LayerNorm (f32 in -> bf16 out), one block per row
// ---------------------------------------------------------------------------
__global__ void layernorm_kernel(const float* __restrict__ X,
                                 const float* __restrict__ w,
                                 const float* __restrict__ b,
                                 unsigned short* __restrict__ Y, int R) {
    __shared__ float red[256];
    int row = blockIdx.x;
    const float* xr = X + (size_t)row * R;
    float s = 0.f;
    for (int c = threadIdx.x; c < R; c += 256) s += xr[c];
    red[threadIdx.x] = s; __syncthreads();
    for (int off = 128; off > 0; off >>= 1) {
        if (threadIdx.x < off) red[threadIdx.x] += red[threadIdx.x + off];
        __syncthreads();
    }
    float mu = red[0] / (float)R;
    __syncthreads();
    float v = 0.f;
    for (int c = threadIdx.x; c < R; c += 256) { float d = xr[c] - mu; v += d * d; }
    red[threadIdx.x] = v; __syncthreads();
    for (int off = 128; off > 0; off >>= 1) {
        if (threadIdx.x < off) red[threadIdx.x] += red[threadIdx.x + off];
        __syncthreads();
    }
    float rstd = rsqrtf(red[0] / (float)R + 1e-5f);
    for (int c = threadIdx.x; c < R; c += 256)
        Y[(size_t)row * R + c] = f2bf((xr[c] - mu) * rstd * w[c] + b[c]);
}

// ---------------------------------------------------------------------------
// 4) WMMA GEMM:  Out[M,N] = A[M,K] (bf16, row-major) * W[N,K]^T (bf16)
//    grid = (ceil(tilesN/8), M/16); 8 waves/block, one 16x16 tile per wave.
//    The 16xK A tile (shared by the whole block) is DMA'd into LDS by the
//    Tensor Data Mover (tensor_load_to_lds, TENSORcnt), then read as
//    ds_load fragments; B (weights) streams from global (L2-resident) with
//    global_prefetch_b8. Fragment packing per CDNA5 ISA 7.12.2 (wave32).
// ---------------------------------------------------------------------------
__global__ void wmma_gemm_bf16_kernel(const unsigned short* __restrict__ A,
                                      const unsigned short* __restrict__ Bw,
                                      float* __restrict__ Out,
                                      int M, int N, int K) {
    __shared__ unsigned short tileA[16 * 384];      // K <= 384 -> 12 KB
    const int tm = blockIdx.y;                      // row tile
    const int wv = threadIdx.x >> 5;                // wave in block (0..7)
    const int tn = blockIdx.x * 8 + wv;             // col tile
    const int lane = threadIdx.x & 31;
    const int tilesN = N >> 4;

    if (wv == 0) {
        // --- Tensor DMA descriptor (D#), CDNA5 ISA ch. 8 ---
        unsigned long long ga = (unsigned long long)(const void*)(A + (size_t)tm * 16 * K);
        unsigned lds  = (unsigned)(unsigned long long)(const void*)tileA; // LDS offset (addr[31:0])
        unsigned td0  = (unsigned)(K >> 1);          // row length in dwords
        v4u g0;
        g0[0] = 1u;                                  // count=1, no gather
        g0[1] = lds;                                 // lds_addr
        g0[2] = (unsigned)(ga & 0xFFFFFFFFu);        // global_addr[31:0]
        g0[3] = (unsigned)((ga >> 32) & 0x01FFFFFFu) // global_addr[56:32]
              | (2u << 30);                          // type = 2 (image)
        v8u g1;
        g1[0] = 2u << 16;                            // data_size = 4 bytes
        g1[1] = (td0 & 0xFFFFu) << 16;               // tensor_dim0[15:0]
        g1[2] = (td0 >> 16) | (16u << 16);           // tensor_dim0[31:16] | tensor_dim1=16
        g1[3] = (td0 << 16);                         // tensor_dim1 hi=0 | tile_dim0
        g1[4] = 16u;                                 // tile_dim1=16, tile_dim2=0
        g1[5] = td0;                                 // tensor_dim0_stride[31:0]
        g1[6] = 0u;                                  // stride hi | dim1_stride lo
        g1[7] = 0u;
        asm volatile("tensor_load_to_lds %0, %1" :: "s"(g0), "s"(g1) : "memory");
        __builtin_amdgcn_s_wait_tensorcnt((short)0);
    }
    __syncthreads();

    if (tn < tilesN) {                               // wave-uniform (EXEC stays full)
        const int m  = lane & 15;                    // M for A-frag, N for B-frag/C
        const int hi = lane >> 4;                    // lane-half selector
        const unsigned short* arow = tileA + m * K;  // LDS
        const unsigned short* brow = Bw + (size_t)(tn * 16 + m) * K;

        v8f acc = {};
        for (int k0 = 0; k0 < K; k0 += 32) {
            if (k0 + 32 < K)                         // -> global_prefetch_b8
                __builtin_prefetch(brow + k0 + 32, 0, 3);
            v8u ap, bp;
#pragma unroll
            for (int i = 0; i < 8; ++i) {
                // A 16x32 bf16: lanes<16 -> K 0..7 (v0-3), 16..23 (v4-7); lanes>=16: +8
                int ka = k0 + ((i & 4) << 2) + (hi << 3) + ((i & 3) << 1);
                ap[i] = *(const unsigned int*)(arow + ka);
                // B 32x16 bf16: same K layout as A: lanes<16 -> K 0..7 (v0-3), 16..23 (v4-7); lanes>=16: +8
                int kb = k0 + ((i & 4) << 2) + (hi << 3) + ((i & 3) << 1);
                bp[i] = *(const unsigned int*)(brow + kb);
            }
            v16bf av = __builtin_bit_cast(v16bf, ap);
            v16bf bv = __builtin_bit_cast(v16bf, bp);
            acc = __builtin_amdgcn_wmma_f32_16x16x32_bf16(
                      false, av, false, bv, (short)0, acc, false, false);
        }
        // C/D layout: VGPR i -> M = hi*8 + i, N = lane&15
        float* orow = Out + (size_t)(tm * 16 + hi * 8) * N + tn * 16 + m;
#pragma unroll
        for (int i = 0; i < 8; ++i) orow[(size_t)i * N] = acc[i];
    }
}

// ---------------------------------------------------------------------------
// 5) depthwise conv1d (k=3, pad=1) over L within each batch + SiLU
// ---------------------------------------------------------------------------
__global__ void conv1d_silu_kernel(const float* __restrict__ xz,
                                   const float* __restrict__ cw,   // (384,1,3)
                                   const float* __restrict__ cb,
                                   float* __restrict__ xcf,
                                   unsigned short* __restrict__ xcb) {
    int idx = blockIdx.x * 256 + threadIdx.x;          // NROW*DIN
    if (idx >= NROW * DIN) return;
    int e = idx % DIN;
    int row = idx / DIN;
    int l = row % LS;
    const float* base = xz + (size_t)row * DXZ + e;    // x_in = xz[:, :384]
    float s = cb[e];
    float w0 = cw[e * 3 + 0], w1 = cw[e * 3 + 1], w2 = cw[e * 3 + 2];
    if (l > 0)      s += base[-DXZ] * w0;
    s += base[0] * w1;
    if (l < LS - 1) s += base[DXZ]  * w2;
    float v = s / (1.f + __expf(-s));                  // SiLU
    xcf[idx] = v;
    xcb[idx] = f2bf(v);
}

// ---------------------------------------------------------------------------
// 6) delta = softplus(dbc[...,0] * dt_w[e] + dt_b[e])
// ---------------------------------------------------------------------------
__global__ void delta_kernel(const float* __restrict__ dbc,
                             const float* __restrict__ dtw,   // (384,1)
                             const float* __restrict__ dtb,
                             float* __restrict__ delta) {
    int idx = blockIdx.x * 256 + threadIdx.x;          // NROW*DIN
    if (idx >= NROW * DIN) return;
    int e = idx % DIN;
    int row = idx / DIN;
    float pre = dbc[(size_t)row * DBCP] * dtw[e] + dtb[e];
    delta[idx] = (pre > 20.f) ? pre : log1pf(__expf(pre));
}

// ---------------------------------------------------------------------------
// 7) selective scan: one lane per (b, e, n) state; 16-lane shfl reduction
// ---------------------------------------------------------------------------
__global__ void scan_kernel(const float* __restrict__ delta,
                            const float* __restrict__ dbc,
                            const float* __restrict__ xcf,
                            const float* __restrict__ xz,
                            const float* __restrict__ A_log,  // (384,16)
                            const float* __restrict__ Dv,     // (384,)
                            float* __restrict__ y) {
    int g = blockIdx.x * 256 + threadIdx.x;            // BSZ*DIN*DST = 196608
    if (g >= BSZ * DIN * DST) return;
    int n   = g & 15;
    int seq = g >> 4;                                  // (b, e)
    int e = seq % DIN;
    int b = seq / DIN;
    float Aen = -__expf(A_log[e * DST + n]);
    float De  = Dv[e];
    float h = 0.f;
    int row = b * LS;
    for (int l = 0; l < LS; ++l, ++row) {
        float d  = delta[(size_t)row * DIN + e];
        float xc = xcf  [(size_t)row * DIN + e];
        float bn = dbc  [(size_t)row * DBCP + 1 + n];
        float cn = dbc  [(size_t)row * DBCP + 17 + n];
        h = __expf(d * Aen) * h + d * bn * xc;
        float val = h * cn;
        val += __shfl_xor(val, 1, 16);
        val += __shfl_xor(val, 2, 16);
        val += __shfl_xor(val, 4, 16);
        val += __shfl_xor(val, 8, 16);
        if (n == 0) {
            float zv  = xz[(size_t)row * DXZ + DIN + e];
            float sil = zv / (1.f + __expf(-zv));
            y[(size_t)row * DIN + e] = (val + xc * De) * sil;
        }
    }
}

// ---------------------------------------------------------------------------
// 8) final: bilinear 16->64 upsample + LDS-tiled depthwise 7x7 + residual
// ---------------------------------------------------------------------------
__global__ void final_kernel(const float* __restrict__ x,
                             const float* __restrict__ ssm_small,  // (B, Ls, C)
                             const float* __restrict__ dww,        // (192,1,7,7)
                             const float* __restrict__ dwb,
                             const float* __restrict__ gate,
                             float* __restrict__ out) {
    __shared__ float tile[22][23];
    __shared__ float wsh[49];
    const int c = blockIdx.z % CDIM, b = blockIdx.z / CDIM;
    const int tx = threadIdx.x, ty = threadIdx.y;
    const int w0 = blockIdx.x * 16, h0 = blockIdx.y * 16;
    const float* xp = x + (size_t)(b * CDIM + c) * (HDIM * WDIM);

    for (int t = ty * 16 + tx; t < 22 * 22; t += 256) {
        int i = t / 22, j = t % 22;
        int hh = h0 - 3 + i, ww = w0 - 3 + j;
        tile[i][j] = (hh >= 0 && hh < HDIM && ww >= 0 && ww < WDIM)
                         ? xp[hh * WDIM + ww] : 0.f;
    }
    int t0 = ty * 16 + tx;
    if (t0 < 49) wsh[t0] = dww[c * 49 + t0];
    __syncthreads();

    int h = h0 + ty, w = w0 + tx;
    float local = dwb[c];
#pragma unroll
    for (int i = 0; i < 7; ++i)
#pragma unroll
        for (int j = 0; j < 7; ++j)
            local += tile[ty + i][tx + j] * wsh[i * 7 + j];

    // bilinear, half-pixel centers, clamped (16 -> 64, scale 4)
    float sh = (h + 0.5f) * 0.25f - 0.5f;
    float sw = (w + 0.5f) * 0.25f - 0.5f;
    int i0 = (int)floorf(sh); float fh = sh - (float)i0;
    int j0 = (int)floorf(sw); float fw = sw - (float)j0;
    int i1 = min(15, max(0, i0 + 1)); i0 = min(15, max(0, i0));
    int j1 = min(15, max(0, j0 + 1)); j0 = min(15, max(0, j0));
    const float* sp = ssm_small + (size_t)b * LS * CDIM + c;
    float v00 = sp[(i0 * 16 + j0) * CDIM], v01 = sp[(i0 * 16 + j1) * CDIM];
    float v10 = sp[(i1 * 16 + j0) * CDIM], v11 = sp[(i1 * 16 + j1) * CDIM];
    float ssm = v00 * (1.f - fh) * (1.f - fw) + v01 * (1.f - fh) * fw +
                v10 * fh * (1.f - fw)         + v11 * fh * fw;

    float g = 1.f / (1.f + __expf(-gate[0]));
    out[(size_t)(b * CDIM + c) * (HDIM * WDIM) + h * WDIM + w] =
        xp[h * WDIM + w] + g * ssm + (1.f - g) * local;
}

// ---------------------------------------------------------------------------
extern "C" void kernel_launch(void* const* d_in, const int* in_sizes, int n_in,
                              void* d_out, int out_size, void* d_ws, size_t ws_size,
                              hipStream_t stream) {
    const float* x    = (const float*)d_in[0];
    const float* ipw  = (const float*)d_in[1];
    const float* xpw  = (const float*)d_in[2];
    const float* dtw  = (const float*)d_in[3];
    const float* dtb  = (const float*)d_in[4];
    const float* cw   = (const float*)d_in[5];
    const float* cb   = (const float*)d_in[6];
    const float* Alog = (const float*)d_in[7];
    const float* Dv   = (const float*)d_in[8];
    const float* opw  = (const float*)d_in[9];
    const float* nsw  = (const float*)d_in[10];
    const float* nsb  = (const float*)d_in[11];
    const float* n1w  = (const float*)d_in[12];
    const float* n1b  = (const float*)d_in[13];
    const float* dww  = (const float*)d_in[14];
    const float* dwb  = (const float*)d_in[15];
    const float* gate = (const float*)d_in[16];
    float* out = (float*)d_out;

    char* ws = (char*)d_ws;
    size_t off = 0;
    auto alloc = [&](size_t bytes) -> void* {
        void* p = ws + off;
        off = (off + bytes + 255) & ~(size_t)255;
        return p;
    };
    float*          x_flat   = (float*)         alloc((size_t)NROW * CDIM * 4);
    unsigned short* x_ln     = (unsigned short*)alloc((size_t)NROW * CDIM * 2);
    float*          xz       = (float*)         alloc((size_t)NROW * DXZ * 4);
    float*          xcf      = (float*)         alloc((size_t)NROW * DIN * 4);
    unsigned short* xcb      = (unsigned short*)alloc((size_t)NROW * DIN * 2);
    float*          dbc      = (float*)         alloc((size_t)NROW * DBCP * 4);
    float*          delta    = (float*)         alloc((size_t)NROW * DIN * 4);
    float*          yb       = (float*)         alloc((size_t)NROW * DIN * 4);
    unsigned short* y_ln     = (unsigned short*)alloc((size_t)NROW * DIN * 2);
    float*          ssm_s    = (float*)         alloc((size_t)NROW * CDIM * 4);
    unsigned short* ipb      = (unsigned short*)alloc((size_t)DXZ * CDIM * 2);
    unsigned short* xpb      = (unsigned short*)alloc((size_t)DBCP * DIN * 2);
    unsigned short* opb      = (unsigned short*)alloc((size_t)CDIM * DIN * 2);

    // 1) weights -> bf16
    convert_weights_kernel<<<(DXZ * CDIM + 255) / 256, 256, 0, stream>>>(
        ipw, xpw, opw, ipb, xpb, opb);
    // 2) pool 4x4 -> (B,Ls,C)
    pool_kernel<<<(NROW * CDIM + 255) / 256, 256, 0, stream>>>(x, x_flat);
    // 3) LN over C -> bf16
    layernorm_kernel<<<NROW, 256, 0, stream>>>(x_flat, n1w, n1b, x_ln, CDIM);
    // 4) in_proj: (8192x192)*(192x768) -> xz   (tilesN=48 -> grid.x=6)
    wmma_gemm_bf16_kernel<<<dim3((DXZ / 16 + 7) / 8, NROW / 16), 256, 0, stream>>>(
        x_ln, ipb, xz, NROW, DXZ, CDIM);
    // 5) conv1d + SiLU
    conv1d_silu_kernel<<<(NROW * DIN + 255) / 256, 256, 0, stream>>>(
        xz, cw, cb, xcf, xcb);
    // 6) x_proj: (8192x384)*(384x48) -> dbc   (tilesN=3 -> grid.x=1)
    wmma_gemm_bf16_kernel<<<dim3((DBCP / 16 + 7) / 8, NROW / 16), 256, 0, stream>>>(
        xcb, xpb, dbc, NROW, DBCP, DIN);
    // 7) delta = softplus(...)
    delta_kernel<<<(NROW * DIN + 255) / 256, 256, 0, stream>>>(dbc, dtw, dtb, delta);
    // 8) selective scan + gating
    scan_kernel<<<(BSZ * DIN * DST + 255) / 256, 256, 0, stream>>>(
        delta, dbc, xcf, xz, Alog, Dv, yb);
    // 9) LN over d_inner -> bf16
    layernorm_kernel<<<NROW, 256, 0, stream>>>(yb, nsw, nsb, y_ln, DIN);
    // 10) out_proj: (8192x384)*(384x192) -> ssm_small   (tilesN=12 -> grid.x=2)
    wmma_gemm_bf16_kernel<<<dim3((CDIM / 16 + 7) / 8, NROW / 16), 256, 0, stream>>>(
        y_ln, opb, ssm_s, NROW, CDIM, DIN);
    // 11) upsample + depthwise 7x7 + gated residual
    {
        dim3 blk(16, 16);
        dim3 grd(WDIM / 16, HDIM / 16, BSZ * CDIM);
        final_kernel<<<grd, blk, 0, stream>>>(x, ssm_s, dww, dwb, gate, out);
    }
    (void)in_sizes; (void)n_in; (void)out_size; (void)ws_size;
}
